// UnifiedCascadePredictionModel_55319178772966
// MI455X (gfx1250) — matrix operand order repaired and checked by reading, not compiled, not run on hardware
//
#include <hip/hip_runtime.h>
#include <hip/hip_fp16.h>
#include <math.h>

typedef __attribute__((ext_vector_type(16))) _Float16 v16h;
typedef __attribute__((ext_vector_type(8)))  _Float16 v8h;
typedef __attribute__((ext_vector_type(8)))  float    v8f;

#define BB   8
#define NNODE 2048
#define CC   256            // input channels (K of GEMM)
#define EE   32768
#define HH   8
#define CO   64
#define HC   (HH * CO)      // 512 (N of GEMM)
#define BN   (BB * NNODE)   // 16384 (M of GEMM)
#define ETOT (BB * EE + BN) // 278528 edges incl. self loops
#define NEG_SLOPE 0.2f

// ---------------- helpers ----------------

__device__ __forceinline__ unsigned fkey(float f) {
    unsigned u = __float_as_uint(f);
    return (u & 0x80000000u) ? ~u : (u | 0x80000000u);
}
__device__ __forceinline__ float funkey(unsigned u) {
    u = (u & 0x80000000u) ? (u & 0x7fffffffu) : ~u;
    return __uint_as_float(u);
}

// edge id -> (src, dst) node ids in [0, BN)
__device__ __forceinline__ void edge_sd(const int* __restrict__ ei, int e, int& s, int& d) {
    if (e < BB * EE) {
        int b = e / EE;
        int r = e - b * EE;
        int o = b * NNODE;
        s = ei[r] + o;
        d = ei[EE + r] + o;
    } else {
        int n = e - BB * EE;
        s = n; d = n;
    }
}

// ---------------- prep kernels ----------------

// fp32 -> fp16 copy of x
__global__ __launch_bounds__(256) void cvt_x_kernel(const float* __restrict__ x,
                                                    _Float16* __restrict__ xh, int n) {
    int i = blockIdx.x * blockDim.x + threadIdx.x;
    if (i < n) xh[i] = (_Float16)x[i];
}

// transpose + convert W[CC][HC] -> whT[HC][CC] (f16) so B-fragments load contiguous K
__global__ __launch_bounds__(256) void cvt_w_kernel(const float* __restrict__ w,
                                                    _Float16* __restrict__ whT) {
    int i = blockIdx.x * blockDim.x + threadIdx.x;   // i = n*CC + k
    if (i >= CC * HC) return;
    int n = i >> 8;          // / CC
    int k = i & (CC - 1);
    whT[i] = (_Float16)w[(size_t)k * HC + n];
}

// init: out = bias broadcast; mkey = 0 (== -inf key); ssum = 0
__global__ __launch_bounds__(256) void init_kernel(const float* __restrict__ bias,
                                                   float* __restrict__ out,
                                                   unsigned* __restrict__ mkey,
                                                   float* __restrict__ ssum) {
    int i = blockIdx.x * blockDim.x + threadIdx.x;
    if (i < BN * HC) out[i] = bias[i & (HC - 1)];
    if (i < BN * HH) { mkey[i] = 0u; ssum[i] = 0.0f; }
}

// ---------------- WMMA projection: xt = x @ W  (f16 in, f32 out) ----------------
// Each wave: 16 rows x 64 cols, K looped 0..256 step 32. 8 waves/block -> 128 rows.
__global__ __launch_bounds__(256) void gemm_wmma_kernel(const _Float16* __restrict__ xh,
                                                        const _Float16* __restrict__ whT,
                                                        float* __restrict__ xt) {
    const int lane = threadIdx.x & 31;
    const int wave = threadIdx.x >> 5;
    const int m0 = blockIdx.x * 128 + wave * 16;
    const int n0 = blockIdx.y * 64;
    const int r  = lane & 15;   // row (A) / col (B) within tile
    const int hi = lane >> 4;   // lane-group select

    const _Float16* arow = xh + (size_t)(m0 + r) * CC;

    v8f acc[4] = {v8f{}, v8f{}, v8f{}, v8f{}};

    for (int k0 = 0; k0 < CC; k0 += 32) {
        // A fragment: lane-group hi holds K = k0 + hi*16 .. +15 (matches B)
        v16h a = *(const v16h*)(arow + k0 + hi * 16);

        #pragma unroll
        for (int t = 0; t < 4; ++t) {
            // B fragment: column n = n0 + t*16 + r, lane-group hi holds K = k0 + hi*16 .. +15
            const _Float16* bcol = whT + (size_t)(n0 + t * 16 + r) * CC + k0 + hi * 16;
            v16h b = *(const v16h*)bcol;
            acc[t] = __builtin_amdgcn_wmma_f32_16x16x32_f16(
                false, a, false, b, (short)0, acc[t], false, false);
        }
    }

    // C/D layout: VGPR q -> row m0 + hi*8 + q, col n0 + t*16 + r
    const int row = m0 + hi * 8;
    #pragma unroll
    for (int t = 0; t < 4; ++t) {
        const int col = n0 + t * 16 + r;
        #pragma unroll
        for (int q = 0; q < 8; ++q)
            xt[(size_t)(row + q) * HC + col] = acc[t][q];
    }
}

// ---------------- per-node attention logits ----------------
__global__ __launch_bounds__(256) void alpha_kernel(const float* __restrict__ xt,
                                                    const float* __restrict__ att_src,
                                                    const float* __restrict__ att_dst,
                                                    float* __restrict__ as_,
                                                    float* __restrict__ ad_) {
    int i = blockIdx.x * blockDim.x + threadIdx.x;   // i = n*HH + h
    if (i >= BN * HH) return;
    int n = i >> 3;
    int h = i & 7;
    const float* v = xt + (size_t)n * HC + h * CO;
    const float* a = att_src + h * CO;
    const float* d = att_dst + h * CO;
    float sa = 0.0f, sd = 0.0f;
    #pragma unroll 8
    for (int c = 0; c < CO; ++c) { sa += v[c] * a[c]; sd += v[c] * d[c]; }
    as_[i] = sa;
    ad_[i] = sd;
}

// ---------------- edge softmax passes ----------------

// pass A: segment max of leaky-relu logits per (dst, h)
__global__ __launch_bounds__(256) void edge_max_kernel(const int* __restrict__ ei,
                                                       const float* __restrict__ as_,
                                                       const float* __restrict__ ad_,
                                                       unsigned* __restrict__ mkey) {
    int i = blockIdx.x * blockDim.x + threadIdx.x;
    if (i >= ETOT * HH) return;
    int e = i >> 3, h = i & 7;
    int s, d; edge_sd(ei, e, s, d);
    float l = as_[s * HH + h] + ad_[d * HH + h];
    l = (l > 0.0f) ? l : NEG_SLOPE * l;
    atomicMax(&mkey[d * HH + h], fkey(l));
}

// pass B: exp(logit - max), store per-edge, accumulate segment sum
__global__ __launch_bounds__(256) void edge_exp_kernel(const int* __restrict__ ei,
                                                       const float* __restrict__ as_,
                                                       const float* __restrict__ ad_,
                                                       const unsigned* __restrict__ mkey,
                                                       float* __restrict__ expw,
                                                       float* __restrict__ ssum) {
    int i = blockIdx.x * blockDim.x + threadIdx.x;
    if (i >= ETOT * HH) return;
    int e = i >> 3, h = i & 7;
    int s, d; edge_sd(ei, e, s, d);
    float l = as_[s * HH + h] + ad_[d * HH + h];
    l = (l > 0.0f) ? l : NEG_SLOPE * l;
    float ex = expf(l - funkey(mkey[d * HH + h]));
    expw[i] = ex;
    atomicAdd(&ssum[d * HH + h], ex);
}

// pass C: weighted scatter-add of messages; one wave32 per edge, lanes stride the
// 512-float message so loads and atomics are lane-consecutive (coalesced).
__global__ __launch_bounds__(256) void scatter_kernel(const int* __restrict__ ei,
                                                      const float* __restrict__ xt,
                                                      const float* __restrict__ expw,
                                                      const float* __restrict__ ssum,
                                                      float* __restrict__ out) {
    int wid  = (int)((blockIdx.x * (unsigned)blockDim.x + threadIdx.x) >> 5);
    int lane = threadIdx.x & 31;
    if (wid >= ETOT) return;
    int s, d; edge_sd(ei, wid, s, d);
    const float* xs = xt + (size_t)s * HC;
    float*       od = out + (size_t)d * HC;
    #pragma unroll
    for (int it = 0; it < 16; ++it) {
        int j = it * 32 + lane;          // j = h*64 + c
        int h = j >> 6;
        float w = expw[(size_t)wid * HH + h] / (ssum[(size_t)d * HH + h] + 1e-16f);
        atomicAdd(&od[j], w * xs[j]);
    }
}

// ---------------- launch ----------------

extern "C" void kernel_launch(void* const* d_in, const int* in_sizes, int n_in,
                              void* d_out, int out_size, void* d_ws, size_t ws_size,
                              hipStream_t stream) {
    const float* x       = (const float*)d_in[0];
    const int*   ei      = (const int*)d_in[1];    // [2, E] int32
    const float* W_lin   = (const float*)d_in[2];
    const float* att_src = (const float*)d_in[3];
    const float* att_dst = (const float*)d_in[4];
    const float* bias    = (const float*)d_in[5];
    float* out = (float*)d_out;

    char* ws = (char*)d_ws;
    size_t off = 0;
    _Float16* xh   = (_Float16*)(ws + off); off += (size_t)BN * CC * 2;      // 8 MB
    _Float16* whT  = (_Float16*)(ws + off); off += (size_t)HC * CC * 2;      // 256 KB
    float*    xt   = (float*)(ws + off);    off += (size_t)BN * HC * 4;      // 32 MB
    float*    as_  = (float*)(ws + off);    off += (size_t)BN * HH * 4;
    float*    ad_  = (float*)(ws + off);    off += (size_t)BN * HH * 4;
    unsigned* mkey = (unsigned*)(ws + off); off += (size_t)BN * HH * 4;
    float*    ssum = (float*)(ws + off);    off += (size_t)BN * HH * 4;
    float*    expw = (float*)(ws + off);    off += (size_t)ETOT * HH * 4;    // 8.5 MB

    // 1) fp16 copies
    cvt_x_kernel<<<(BN * CC + 255) / 256, 256, 0, stream>>>(x, xh, BN * CC);
    cvt_w_kernel<<<(CC * HC + 255) / 256, 256, 0, stream>>>(W_lin, whT);

    // 2) init output (bias) + softmax state
    init_kernel<<<(BN * HC + 255) / 256, 256, 0, stream>>>(bias, out, mkey, ssum);

    // 3) WMMA projection xt = x @ W
    dim3 ggrid(BN / 128, HC / 64);
    gemm_wmma_kernel<<<ggrid, 256, 0, stream>>>(xh, whT, xt);

    // 4) per-node logits
    alpha_kernel<<<(BN * HH + 255) / 256, 256, 0, stream>>>(xt, att_src, att_dst, as_, ad_);

    // 5) segment softmax + scatter
    int ethreads = ETOT * HH;
    edge_max_kernel<<<(ethreads + 255) / 256, 256, 0, stream>>>(ei, as_, ad_, mkey);
    edge_exp_kernel<<<(ethreads + 255) / 256, 256, 0, stream>>>(ei, as_, ad_, mkey, expw, ssum);
    scatter_kernel<<<(ETOT * 32 + 255) / 256, 256, 0, stream>>>(ei, xt, expw, ssum, out);
    (void)in_sizes; (void)n_in; (void)out_size; (void)ws_size;
}
